// SSBlock_84559316123650
// MI455X (gfx1250) — matrix operand order repaired and checked by reading, not compiled, not run on hardware
//
#include <hip/hip_runtime.h>

// ---------------------------------------------------------------------------
// CDNA5 (gfx1250) WMMA types & helpers
// ---------------------------------------------------------------------------
typedef __attribute__((ext_vector_type(16))) __bf16 v16bf;
typedef __attribute__((ext_vector_type(8)))  float  v8f;

union FragB {
  uint4  q[2];
  __bf16 h[16];
  v16bf  v;
};

__device__ __forceinline__ v8f v8f_zero() {
  v8f z = {0.f, 0.f, 0.f, 0.f, 0.f, 0.f, 0.f, 0.f};
  return z;
}

__device__ __forceinline__ v8f wmma_bf16(v16bf a, v16bf b, v8f c) {
  // D = A(16x32) * B(32x16) + C, f32 accumulate
  return __builtin_amdgcn_wmma_f32_16x16x32_bf16(false, a, false, b, (short)0, c,
                                                 false, false);
}

// A fragment (16x32 MxK) from row-major bf16 [m][k], row stride ld elements.
// ISA layout: lane m = lane&15; elements e<8 -> k = hi8+e, e>=8 -> k = 16+hi8+(e-8)
// where hi8 = (lane>=16)*8.  Two 16B loads per lane.
__device__ __forceinline__ v16bf load_a_rm(const __bf16* base, int ld) {
  int lane = threadIdx.x & 31;
  const __bf16* p = base + (lane & 15) * ld + ((lane >> 4) << 3);
  FragB f;
  f.q[0] = *(const uint4*)(p);
  f.q[1] = *(const uint4*)(p + 16);
  return f.v;
}

// Same A fragment, but source is f32; convert to bf16 while building fragment.
__device__ __forceinline__ v16bf load_a_rm_f32(const float* base, int ld) {
  int lane = threadIdx.x & 31;
  const float* p = base + (lane & 15) * ld + ((lane >> 4) << 3);
  FragB f;
#pragma unroll
  for (int i = 0; i < 8; ++i) f.h[i] = (__bf16)p[i];
#pragma unroll
  for (int i = 0; i < 8; ++i) f.h[8 + i] = (__bf16)p[16 + i];
  return f.v;
}

// B fragment (32x16 KxN) from "k-contiguous" storage Bt[n][k], row stride ld.
// ISA layout: lane n = lane&15; elements e<8 -> k = hi8+e, e>=8 -> k = 16+hi8+(e-8)
// where hi8 = (lane>=16)*8. Two 16B loads per lane.
__device__ __forceinline__ v16bf load_b_kc(const __bf16* base, int ld) {
  int lane = threadIdx.x & 31;
  const __bf16* p = base + (lane & 15) * ld + ((lane >> 4) << 3);
  FragB f;
  f.q[0] = *(const uint4*)(p);
  f.q[1] = *(const uint4*)(p + 16);
  return f.v;
}

// ---------------------------------------------------------------------------
// Geometry constants
// ---------------------------------------------------------------------------
#define BATCH 2
#define CH    128
#define HH    192
#define WW    192
#define FF    8        // unshuffle factor
#define SH    24       // 192/8
#define NTOK  576      // 24*24 tokens per sub-image
#define NSUB  128      // B * f * f
#define CHW   (CH * HH * WW)

// ---------------------------------------------------------------------------
// 1) LayerNorm statistics (sum, sumsq) per batch, atomic partial reduction
// ---------------------------------------------------------------------------
__global__ void stats_partial_kernel(const float* __restrict__ x, float* __restrict__ stats) {
  int b = blockIdx.y;
  const float* xb = x + (size_t)b * CHW;
  float s = 0.f, sq = 0.f;
  for (int i = blockIdx.x * blockDim.x + threadIdx.x; i < CHW; i += gridDim.x * blockDim.x) {
    float v = xb[i];
    s += v;
    sq += v * v;
  }
#pragma unroll
  for (int off = 16; off; off >>= 1) {
    s += __shfl_xor(s, off, 32);
    sq += __shfl_xor(sq, off, 32);
  }
  __shared__ float ls[8], lq[8];
  int wid = threadIdx.x >> 5;
  if ((threadIdx.x & 31) == 0) { ls[wid] = s; lq[wid] = sq; }
  __syncthreads();
  if (threadIdx.x == 0) {
    float S = 0.f, Q = 0.f;
    for (int w = 0; w < 8; ++w) { S += ls[w]; Q += lq[w]; }
    atomicAdd(&stats[b * 2 + 0], S);
    atomicAdd(&stats[b * 2 + 1], Q);
  }
}

__global__ void stats_finalize_kernel(float* __restrict__ stats) {
  if (threadIdx.x < BATCH) {
    int b = threadIdx.x;
    float mean = stats[b * 2 + 0] / (float)CHW;
    float var  = stats[b * 2 + 1] / (float)CHW - mean * mean;
    stats[4 + b] = mean;
    stats[6 + b] = rsqrtf(var + 1e-5f);
  }
}

// ---------------------------------------------------------------------------
// 2) Repack weights to bf16 WMMA-friendly layouts
//    wqkT[e][c]          (k-contiguous B for qk GEMM)
//    wA1/wA2[co][tap*128+ci]  (k-contiguous A for implicit-GEMM convs)
//    wA3[co][ci]
// ---------------------------------------------------------------------------
__global__ void prep_weights_kernel(const float* __restrict__ wqk,
                                    const float* __restrict__ w1,
                                    const float* __restrict__ w2,
                                    const float* __restrict__ w3,
                                    __bf16* __restrict__ wqkT,
                                    __bf16* __restrict__ wA1,
                                    __bf16* __restrict__ wA2,
                                    __bf16* __restrict__ wA3) {
  const int N_QK = CH * CH;          // 16384
  const int N_C  = CH * CH * 9;      // 147456
  const int total = N_QK + 2 * N_C + N_QK;
  for (int i = blockIdx.x * blockDim.x + threadIdx.x; i < total;
       i += gridDim.x * blockDim.x) {
    int t = i;
    if (t < N_QK) {
      int e = t >> 7, c = t & 127;
      wqkT[t] = (__bf16)wqk[c * CH + e];
    } else if (t < N_QK + 2 * N_C) {
      int u = t - N_QK;
      const float* w = (u < N_C) ? w1 : w2;
      __bf16* dst = (u < N_C) ? wA1 : wA2;
      if (u >= N_C) u -= N_C;
      int co = u / 1152;
      int r = u - co * 1152;
      int tap = r >> 7, ci = r & 127;
      int kh = tap / 3, kw = tap - kh * 3;
      dst[u] = (__bf16)w[((co * CH + ci) * 3 + kh) * 3 + kw];
    } else {
      int u = t - (N_QK + 2 * N_C);
      wA3[u] = (__bf16)w3[u];
    }
  }
}

// ---------------------------------------------------------------------------
// 3) Unshuffle x into:
//    Qln[s][n][c] = bf16(layernorm(x))   (row-major, k-contiguous for A frags)
//    Vt [s][c][n] = bf16(x)              (key-contiguous for PV B frags)
//    s = b*64 + dy*8 + dx ; n = i*24 + j ; src = x[b][c][i*8+dy][j*8+dx]
// ---------------------------------------------------------------------------
__global__ void qkv_prep_kernel(const float* __restrict__ x,
                                const float* __restrict__ stats,
                                __bf16* __restrict__ Qln,
                                __bf16* __restrict__ Vt) {
  const int total = BATCH * CHW;
  for (int idx = blockIdx.x * blockDim.x + threadIdx.x; idx < total;
       idx += gridDim.x * blockDim.x) {
    int b = idx / CHW;
    int r = idx - b * CHW;
    int c = r / (HH * WW);
    int hw = r - c * (HH * WW);
    int h = hw / WW, w = hw - h * WW;
    int dy = h & 7, ii = h >> 3;
    int dx = w & 7, jj = w >> 3;
    int s = (b << 6) + (dy << 3) + dx;
    int n = ii * SH + jj;
    float v = x[idx];
    float q = (v - stats[4 + b]) * stats[6 + b];
    Qln[((size_t)s * NTOK + n) * CH + c] = (__bf16)q;
    Vt[((size_t)s * CH + c) * NTOK + n] = (__bf16)v;
  }
}

// ---------------------------------------------------------------------------
// 4) qk = Qln @ wqk  (576x128 per sub-image, K=128), then row-L2-normalize,
//    write K-hat bf16 IN PLACE over Qln (each block owns its 64 rows).
// ---------------------------------------------------------------------------
__global__ __launch_bounds__(256) void qk_norm_gemm_kernel(
    const __bf16* __restrict__ Qln, const __bf16* __restrict__ wqkT,
    __bf16* __restrict__ Khat) {
  __shared__ float QK[64][128];   // 32 KB
  __shared__ float rnorm[64];
  int s = blockIdx.y;
  int n0 = blockIdx.x * 64;
  const __bf16* Q = Qln + ((size_t)s * NTOK + n0) * CH;
  int wave = threadIdx.x >> 5, lane = threadIdx.x & 31;
  int nt = wave;  // column tile 0..7 (E = 128)
  v8f acc[4];
#pragma unroll
  for (int mt = 0; mt < 4; ++mt) acc[mt] = v8f_zero();
#pragma unroll
  for (int k0 = 0; k0 < 128; k0 += 32) {
    v16bf bfr = load_b_kc(wqkT + (size_t)nt * 16 * CH + k0, CH);
#pragma unroll
    for (int mt = 0; mt < 4; ++mt) {
      v16bf afr = load_a_rm(Q + (size_t)mt * 16 * CH + k0, CH);
      acc[mt] = wmma_bf16(afr, bfr, acc[mt]);
    }
  }
  int colN = lane & 15, hi8 = (lane >> 4) << 3;
#pragma unroll
  for (int mt = 0; mt < 4; ++mt)
#pragma unroll
    for (int r2 = 0; r2 < 8; ++r2)
      QK[mt * 16 + hi8 + r2][nt * 16 + colN] = acc[mt][r2];
  __syncthreads();
  if (threadIdx.x < 64) {
    float sq = 0.f;
#pragma unroll 4
    for (int e = 0; e < 128; ++e) { float v = QK[threadIdx.x][e]; sq += v * v; }
    rnorm[threadIdx.x] = 1.0f / (sqrtf(sq) + 1e-8f);
  }
  __syncthreads();
  __bf16* outp = Khat + ((size_t)s * NTOK + n0) * CH;
  for (int i = threadIdx.x; i < 64 * 128; i += blockDim.x) {
    int rr = i >> 7, cc = i & 127;
    outp[i] = (__bf16)(QK[rr][cc] * rnorm[rr]);
  }
}

// ---------------------------------------------------------------------------
// 5) Attention: per (sub-image, 16 query rows):
//    S = Qhat @ Khat^T (WMMA -> LDS f32), softmax (scale 1/sqrt(E); +1 cancels),
//    O = P @ V (WMMA, V streamed from Vt), y = x + shuffle(O)   (bf16 out)
//    Q tile staged via CDNA5 async global->LDS DMA (ASYNCcnt path).
// ---------------------------------------------------------------------------
__global__ __launch_bounds__(256) void attn_kernel(
    const float* __restrict__ x, const __bf16* __restrict__ Khat,
    const __bf16* __restrict__ Vt, __bf16* __restrict__ y) {
  __shared__ __bf16 Qs[16][128];  // 4 KB
  __shared__ float  S[16][NTOK];  // 36 KB
  int s = blockIdx.y;
  int n0 = blockIdx.x * 16;
  int b = s >> 6;
  int dy = (s >> 3) & 7, dx = s & 7;
  const __bf16* Krows = Khat + (size_t)s * NTOK * CH;
  {
    // Stage Q tile (rows n0..n0+15 of Khat; Q and K are the same normalized qk)
    // with GLOBAL_LOAD_ASYNC_TO_LDS_B128: 256 lanes x 16B = 4 KB, no VGPR
    // round-trip.  LDS destination = low 32 bits of the flat shared pointer.
    const __bf16* gsrc = Krows + (size_t)n0 * CH + (size_t)threadIdx.x * 8;
    unsigned lds_dst = (unsigned)(size_t)(&Qs[0][0] + (size_t)threadIdx.x * 8);
    asm volatile("global_load_async_to_lds_b128 %0, %1, off"
                 :
                 : "v"(lds_dst), "v"(gsrc)
                 : "memory");
    asm volatile("s_wait_asynccnt 0x0" ::: "memory");
  }
  __syncthreads();
  int wave = threadIdx.x >> 5, lane = threadIdx.x & 31;
  int colN = lane & 15, hi8 = (lane >> 4) << 3;
  // Phase 1: raw dot products into LDS
  for (int nt = wave; nt < 36; nt += 8) {
    v8f acc = v8f_zero();
    const __bf16* Kb = Krows + (size_t)nt * 16 * CH;
#pragma unroll
    for (int k0 = 0; k0 < 128; k0 += 32) {
      v16bf bfr = load_b_kc(Kb + k0, CH);
      v16bf afr = load_a_rm(&Qs[0][0] + k0, CH);
      acc = wmma_bf16(afr, bfr, acc);
    }
#pragma unroll
    for (int r2 = 0; r2 < 8; ++r2) S[hi8 + r2][nt * 16 + colN] = acc[r2];
  }
  __syncthreads();
  // Phase 2: softmax per row; softmax((d+1)*sc) == softmax((d-max)*sc)
  {
    int row = threadIdx.x >> 4, sub = threadIdx.x & 15;
    const float sc = 0.08838834764831845f;  // 1/sqrt(128)
    float mx = -3.0e38f;
    for (int k = sub; k < NTOK; k += 16) mx = fmaxf(mx, S[row][k]);
    mx = fmaxf(mx, __shfl_xor(mx, 1, 16));
    mx = fmaxf(mx, __shfl_xor(mx, 2, 16));
    mx = fmaxf(mx, __shfl_xor(mx, 4, 16));
    mx = fmaxf(mx, __shfl_xor(mx, 8, 16));
    float sum = 0.f;
    for (int k = sub; k < NTOK; k += 16) {
      float p = __expf((S[row][k] - mx) * sc);
      S[row][k] = p;
      sum += p;
    }
    sum += __shfl_xor(sum, 1, 16);
    sum += __shfl_xor(sum, 2, 16);
    sum += __shfl_xor(sum, 4, 16);
    sum += __shfl_xor(sum, 8, 16);
    float inv = 1.0f / sum;
    for (int k = sub; k < NTOK; k += 16) S[row][k] *= inv;
  }
  __syncthreads();
  // Phase 3: O = P @ V ; wave owns channel tile nt = wave (8 * 16 = 128 ch)
  const __bf16* Vs = Vt + (size_t)s * CH * NTOK + (size_t)(wave * 16) * NTOK;
  v8f o = v8f_zero();
  for (int k0 = 0; k0 < NTOK; k0 += 32) {
    v16bf bfr = load_b_kc(Vs + k0, NTOK);
    v16bf afr = load_a_rm_f32(&S[0][0] + k0, NTOK);
    o = wmma_bf16(afr, bfr, o);
  }
  int ch = wave * 16 + colN;
#pragma unroll
  for (int r2 = 0; r2 < 8; ++r2) {
    int qrow = n0 + hi8 + r2;
    int ii = qrow / SH, jj = qrow - ii * SH;
    int h = ii * FF + dy, w = jj * FF + dx;
    size_t xi = (((size_t)b * CH + ch) * HH + h) * WW + w;
    y[xi] = (__bf16)(x[xi] + o[r2]);
  }
}

// ---------------------------------------------------------------------------
// 6) 3x3 dilation-2 pad-2 conv as implicit GEMM (M=co, N=pixels, K=1152).
//    Output tile: 64 co x 64 px (spatial 4x16). Halo (8x20x128 ci-contig bf16,
//    40 KB LDS) staged once; B-fragments are two ds_load_b128 per k-step.
// ---------------------------------------------------------------------------
__global__ __launch_bounds__(256) void conv3x3d2_kernel(
    const __bf16* __restrict__ in, const __bf16* __restrict__ wA,
    const float* __restrict__ bias, __bf16* __restrict__ out, int relu) {
  __shared__ __bf16 halo[8][20][128];  // 40 KB
  int b = blockIdx.z >> 1;
  int co0 = (blockIdx.z & 1) * 64;
  int h0 = blockIdx.y * 4, w0 = blockIdx.x * 16;
  const __bf16* inb = in + (size_t)b * CH * HH * WW;
  for (int p = threadIdx.x; p < 128 * 8; p += blockDim.x) {
    int ci = p >> 3, hr = p & 7;
    int h = h0 + hr - 2;
    bool hok = (h >= 0) && (h < HH);
    const __bf16* srow = inb + ((size_t)ci * HH + (hok ? h : 0)) * WW;
#pragma unroll
    for (int wc = 0; wc < 20; ++wc) {
      int w = w0 + wc - 2;
      __bf16 v = (__bf16)0.f;
      if (hok && w >= 0 && w < WW) v = srow[w];
      halo[hr][wc][ci] = v;
    }
  }
  __syncthreads();
  int wave = threadIdx.x >> 5, lane = threadIdx.x & 31;
  int pt = wave & 3, ct = wave >> 2;  // pixel tile (spatial row), co tiles ct & ct+2
  v8f acc0 = v8f_zero(), acc1 = v8f_zero();
  const __bf16* wrow0 = wA + (size_t)(co0 + ct * 16) * 1152;
  const __bf16* wrow1 = wA + (size_t)(co0 + (ct + 2) * 16) * 1152;
  for (int tap = 0; tap < 9; ++tap) {
    int kh = tap / 3, kw = tap - kh * 3;
    const __bf16* bbase = &halo[pt + 2 * kh][2 * kw][0];  // lane adds pxc*128
    if (tap < 8) {
      // Prefetch next tap's weight rows (global_prefetch_b8) while this tap
      // computes; weights are L2-resident, this hides first-use latency.
      __builtin_prefetch(wrow0 + (size_t)(tap + 1) * 128, 0, 0);
      __builtin_prefetch(wrow1 + (size_t)(tap + 1) * 128, 0, 0);
    }
#pragma unroll
    for (int c0 = 0; c0 < 128; c0 += 32) {
      int k = tap * 128 + c0;
      v16bf bfr = load_b_kc(bbase + c0, 128);
      v16bf a0 = load_a_rm(wrow0 + k, 1152);
      v16bf a1 = load_a_rm(wrow1 + k, 1152);
      acc0 = wmma_bf16(a0, bfr, acc0);
      acc1 = wmma_bf16(a1, bfr, acc1);
    }
  }
  int colN = lane & 15, hi8 = (lane >> 4) << 3;
  int hpix = h0 + pt, wpix = w0 + colN;
#pragma unroll
  for (int m = 0; m < 2; ++m) {
    v8f a = m ? acc1 : acc0;
    int cobase = co0 + (ct + 2 * m) * 16 + hi8;
#pragma unroll
    for (int r2 = 0; r2 < 8; ++r2) {
      float v = a[r2] + bias[cobase + r2];
      if (relu) v = fmaxf(v, 0.f);
      out[(((size_t)b * CH + cobase + r2) * HH + hpix) * WW + wpix] = (__bf16)v;
    }
  }
}

// ---------------------------------------------------------------------------
// 7) 1x1 conv + bias + residual (f32 out): out = x + w3 @ h2 + b3
// ---------------------------------------------------------------------------
__global__ __launch_bounds__(256) void conv1x1_res_kernel(
    const __bf16* __restrict__ in, const __bf16* __restrict__ wA3,
    const float* __restrict__ bias, const float* __restrict__ x,
    float* __restrict__ out) {
  __shared__ __bf16 T[64][128];  // [pixel][ci], 16 KB
  int b = blockIdx.z >> 1;
  int co0 = (blockIdx.z & 1) * 64;
  int h0 = blockIdx.y * 4, w0 = blockIdx.x * 16;
  const __bf16* inb = in + (size_t)b * CH * HH * WW;
  for (int p = threadIdx.x; p < 64 * 128; p += blockDim.x) {
    int px = p & 63, ci = p >> 6;
    int py = px >> 4, pxc = px & 15;
    T[px][ci] = inb[((size_t)ci * HH + h0 + py) * WW + w0 + pxc];
  }
  __syncthreads();
  int wave = threadIdx.x >> 5, lane = threadIdx.x & 31;
  int pt = wave & 3, ct = wave >> 2;
  v8f acc0 = v8f_zero(), acc1 = v8f_zero();
#pragma unroll
  for (int c0 = 0; c0 < 128; c0 += 32) {
    v16bf bfr = load_b_kc(&T[pt * 16][0] + c0, 128);
    v16bf a0 = load_a_rm(wA3 + (size_t)(co0 + ct * 16) * 128 + c0, 128);
    v16bf a1 = load_a_rm(wA3 + (size_t)(co0 + (ct + 2) * 16) * 128 + c0, 128);
    acc0 = wmma_bf16(a0, bfr, acc0);
    acc1 = wmma_bf16(a1, bfr, acc1);
  }
  int colN = lane & 15, hi8 = (lane >> 4) << 3;
  size_t hw = ((size_t)(h0 + pt)) * WW + w0 + colN;
#pragma unroll
  for (int m = 0; m < 2; ++m) {
    v8f a = m ? acc1 : acc0;
    int cobase = co0 + (ct + 2 * m) * 16 + hi8;
#pragma unroll
    for (int r2 = 0; r2 < 8; ++r2) {
      size_t idx = ((size_t)b * CH + cobase + r2) * (size_t)(HH * WW) + hw;
      out[idx] = x[idx] + a[r2] + bias[cobase + r2];
    }
  }
}

// ---------------------------------------------------------------------------
// Host launch
// ---------------------------------------------------------------------------
extern "C" void kernel_launch(void* const* d_in, const int* in_sizes, int n_in,
                              void* d_out, int out_size, void* d_ws, size_t ws_size,
                              hipStream_t stream) {
  (void)in_sizes; (void)n_in; (void)out_size; (void)ws_size;
  const float* x   = (const float*)d_in[0];
  const float* wqk = (const float*)d_in[1];
  const float* w1  = (const float*)d_in[2];
  const float* b1  = (const float*)d_in[3];
  const float* w2  = (const float*)d_in[4];
  const float* b2  = (const float*)d_in[5];
  const float* w3  = (const float*)d_in[6];
  const float* b3  = (const float*)d_in[7];
  float* out = (float*)d_out;

  char* ws = (char*)d_ws;
  size_t off = 0;
  auto alloc = [&](size_t bytes) -> void* {
    void* p = ws + off;
    off = (off + bytes + 255) & ~(size_t)255;
    return p;
  };
  float*  stats = (float*)alloc(64);
  __bf16* wqkT  = (__bf16*)alloc((size_t)CH * CH * 2);
  __bf16* wA1   = (__bf16*)alloc((size_t)CH * CH * 9 * 2);
  __bf16* wA2   = (__bf16*)alloc((size_t)CH * CH * 9 * 2);
  __bf16* wA3   = (__bf16*)alloc((size_t)CH * CH * 2);
  __bf16* Qln   = (__bf16*)alloc((size_t)NSUB * NTOK * CH * 2);  // aliased as Khat
  __bf16* Vt    = (__bf16*)alloc((size_t)NSUB * CH * NTOK * 2);
  __bf16* ybf   = (__bf16*)alloc((size_t)BATCH * CHW * 2);
  __bf16* h1    = (__bf16*)alloc((size_t)BATCH * CHW * 2);
  __bf16* h2    = (__bf16*)alloc((size_t)BATCH * CHW * 2);

  hipMemsetAsync(stats, 0, 64, stream);
  stats_partial_kernel<<<dim3(128, BATCH), 256, 0, stream>>>(x, stats);
  stats_finalize_kernel<<<1, 32, 0, stream>>>(stats);
  prep_weights_kernel<<<1280, 256, 0, stream>>>(wqk, w1, w2, w3, wqkT, wA1, wA2, wA3);
  qkv_prep_kernel<<<9216, 256, 0, stream>>>(x, stats, Qln, Vt);
  qk_norm_gemm_kernel<<<dim3(9, NSUB), 256, 0, stream>>>(Qln, wqkT, Qln);
  attn_kernel<<<dim3(36, NSUB), 256, 0, stream>>>(x, Qln, Vt, ybf);
  conv3x3d2_kernel<<<dim3(12, 48, 4), 256, 0, stream>>>(ybf, wA1, b1, h1, 1);
  conv3x3d2_kernel<<<dim3(12, 48, 4), 256, 0, stream>>>(h1, wA2, b2, h2, 1);
  conv1x1_res_kernel<<<dim3(12, 48, 4), 256, 0, stream>>>(h2, wA3, b3, x, out);
}
